// MultiHeadAttention_35673998360812
// MI455X (gfx1250) — compile-verified, matrix operand order checked by reading, not run on hardware
//
#include <hip/hip_runtime.h>
#include <hip/hip_fp16.h>

// Problem constants
#define BB 4
#define LL 1024
#define HH 16
#define DD 128
#define DM 2048           // HH*DD
#define NQKV 6144         // HH*3*DD
#define KDIM 2048
#define PLD 72            // padded LDS pitch for P tiles (flash)
#define GLD 40            // padded LDS pitch for GEMM tiles (32 + 8)
#define KP  136           // padded LDS pitch for K tile rows (128 + 8)
#define VP  72            // padded LDS pitch for V tile rows (64 + 8)

typedef __attribute__((ext_vector_type(16))) _Float16 v16h;
typedef __attribute__((ext_vector_type(8)))  _Float16 v8h;
typedef __attribute__((ext_vector_type(8)))  float    v8f;

static __device__ __forceinline__ v8f wmma_f32_f16(v16h a, v16h b, v8f c) {
  // 8-arg form: (neg_a, A, neg_b, B, c_mod, C, reuse_a, reuse_b)
  return __builtin_amdgcn_wmma_f32_16x16x32_f16(false, a, false, b, (short)0, c,
                                                false, false);
}

// Async DMA: one 16-byte transfer per lane, global -> LDS (ASYNCcnt tracked).
static __device__ __forceinline__ void async_load_b128(unsigned lds_off,
                                                       const void* gaddr) {
  asm volatile("global_load_async_to_lds_b128 %0, %1, off"
               :: "v"(lds_off), "v"((unsigned long long)gaddr)
               : "memory");
}
static __device__ __forceinline__ void wait_async0() {
  asm volatile("s_wait_asynccnt 0x0" ::: "memory");
}
// Allow the 5 just-issued next-stage transfers to stay in flight (overlap).
static __device__ __forceinline__ void wait_async_le5() {
  asm volatile("s_wait_asynccnt 0x5" ::: "memory");
}

// 16x32 f16 fragment from row-major source (leading dim ld, elements).
// CDNA5 ISA 7.12.2 (16-bit A 16x32): lane L holds row L&15; halves 0..7 hold
// k = off..off+7, halves 8..15 hold k = off+16..off+23, off = (L>=16)?8:0.
static __device__ __forceinline__ v16h load_frag_f16(const _Float16* base,
                                                     int ld, int lane) {
  const int row  = lane & 15;
  const int koff = (lane >> 4) << 3;
  const _Float16* p = base + (size_t)row * ld + koff;
  v8h lo = *(const v8h*)(p);
  v8h hi = *(const v8h*)(p + 16);
  v16h r;
#pragma unroll
  for (int i = 0; i < 8; ++i) { r[i] = lo[i]; r[i + 8] = hi[i]; }
  return r;
}

// ---------------------------------------------------------------------------
// Kernel 0a: elementwise f32 -> f16 convert (for x)
// ---------------------------------------------------------------------------
__global__ __launch_bounds__(256) void k_convert_f16(
    const float* __restrict__ in, _Float16* __restrict__ outp, int n) {
  int i = blockIdx.x * 256 + threadIdx.x;
  if (i < n) outp[i] = (_Float16)in[i];
}

// ---------------------------------------------------------------------------
// Kernel 0b: transpose + convert f32 (K,N) -> f16 (N,K)   (for weights)
// ---------------------------------------------------------------------------
__global__ __launch_bounds__(256) void k_transpose_f16(
    const float* __restrict__ W, _Float16* __restrict__ WT, int K, int N) {
  size_t idx = (size_t)blockIdx.x * blockDim.x + threadIdx.x;
  size_t total = (size_t)K * N;
  if (idx >= total) return;
  int n = (int)(idx % (size_t)N);
  int k = (int)(idx / (size_t)N);
  WT[(size_t)n * K + k] = (_Float16)W[idx];
}

// ---------------------------------------------------------------------------
// Kernel 1/4: GEMM  C(M x N) = A(M x 2048, f16) * BT(N x 2048, f16)^T + bias
// Block: 256 threads = 8 waves; block tile 256(M) x 64(N); k-chunk 32.
// Double-buffered LDS staging via global_load_async_to_lds_b128; each wave
// computes a 32x64 strip = 8 WMMA tiles per k-step.
// ---------------------------------------------------------------------------
static __device__ __forceinline__ void gemm_issue_stage(
    const _Float16* A, const _Float16* BT, int mblk, int nbase, int k,
    unsigned ldsA0, unsigned ldsB0, int t) {
  // A tile: 256 rows x 64B = 1024 b128 -> 4 per thread
#pragma unroll
  for (int i = 0; i < 4; ++i) {
    const int idx = t + 256 * i;
    const int row = idx >> 2, seg = idx & 3;
    async_load_b128(ldsA0 + (unsigned)(row * GLD + seg * 8) * 2,
                    A + (size_t)(mblk + row) * KDIM + k + seg * 8);
  }
  // B tile: 64 rows x 64B = 256 b128 -> 1 per thread
  const int row = t >> 2, seg = t & 3;
  async_load_b128(ldsB0 + (unsigned)(row * GLD + seg * 8) * 2,
                  BT + (size_t)(nbase + row) * KDIM + k + seg * 8);
}

__global__ __launch_bounds__(256) void k_gemm_f16(
    const _Float16* __restrict__ A, const _Float16* __restrict__ BT,
    const float* __restrict__ bias, float* __restrict__ C, int N) {
  const int t     = threadIdx.x;
  const int lane  = t & 31;
  const int wave  = t >> 5;
  const int mblk  = blockIdx.x * 256;
  const int nbase = blockIdx.y * 64;

  __shared__ alignas(16) _Float16 ldsA[2][256 * GLD];
  __shared__ alignas(16) _Float16 ldsB[2][64 * GLD];
  const unsigned ldsA0[2] = {(unsigned)(size_t)(const void*)&ldsA[0][0],
                             (unsigned)(size_t)(const void*)&ldsA[1][0]};
  const unsigned ldsB0[2] = {(unsigned)(size_t)(const void*)&ldsB[0][0],
                             (unsigned)(size_t)(const void*)&ldsB[1][0]};

  v8f acc0[4] = {}, acc1[4] = {};

  gemm_issue_stage(A, BT, mblk, nbase, 0, ldsA0[0], ldsB0[0], t);
  for (int s = 0; s < KDIM / 32; ++s) {
    const int nxt = s + 1;
    if (nxt < KDIM / 32) {
      gemm_issue_stage(A, BT, mblk, nbase, nxt * 32, ldsA0[nxt & 1],
                       ldsB0[nxt & 1], t);
      wait_async_le5();           // stage s landed; stage s+1 still in flight
    } else {
      wait_async0();
    }
    __syncthreads();
    const _Float16* bufA = &ldsA[s & 1][0];
    const _Float16* bufB = &ldsB[s & 1][0];
    v16h a0 = load_frag_f16(bufA + (wave * 32) * GLD, GLD, lane);
    v16h a1 = load_frag_f16(bufA + (wave * 32 + 16) * GLD, GLD, lane);
#pragma unroll
    for (int j = 0; j < 4; ++j) {
      v16h b = load_frag_f16(bufB + (16 * j) * GLD, GLD, lane);
      acc0[j] = wmma_f32_f16(a0, b, acc0[j]);
      acc1[j] = wmma_f32_f16(a1, b, acc1[j]);
    }
    __syncthreads();              // buffer s free for stage s+2 DMA
  }

  const int col0   = nbase + (lane & 15);
  const int rowoff = (lane >> 4) << 3;
#pragma unroll
  for (int sm = 0; sm < 2; ++sm) {
    const int mbase = mblk + wave * 32 + sm * 16;
    const v8f* acc = sm ? acc1 : acc0;
#pragma unroll
    for (int j = 0; j < 4; ++j) {
      const int col = col0 + 16 * j;
      const float bv = bias[col];
#pragma unroll
      for (int r = 0; r < 8; ++r)
        C[(size_t)(mbase + rowoff + r) * N + col] = acc[j][r] + bv;
    }
  }
}

// ---------------------------------------------------------------------------
// Kernel 2: RoPE on q,k + repack to f16.
// qkv f32 (B,L,H,384) -> Qh,Kh f16 (B,H,L,128) and Vt f16 (B,H,128,L)
// ---------------------------------------------------------------------------
__global__ __launch_bounds__(256) void k_rope_pack(
    const float* __restrict__ qkv, const float* __restrict__ cosv,
    const float* __restrict__ sinv, _Float16* __restrict__ Qh,
    _Float16* __restrict__ Kh, _Float16* __restrict__ Vt) {
  int idx = blockIdx.x * blockDim.x + threadIdx.x;   // (b,h,l,d) flat
  if (idx >= BB * HH * LL * DD) return;
  const int d = idx & (DD - 1);
  const int l = (idx >> 7) & (LL - 1);
  const int h = (idx >> 17) & (HH - 1);
  const int b = idx >> 21;
  const float* base = qkv + ((size_t)(b * LL + l) * HH + h) * 384;
  const float c = cosv[l], s = sinv[l];
  float q, k;
  if (d < 64) {
    q = base[d] * c - base[d + 64] * s;
    k = base[128 + d] * c - base[128 + d + 64] * s;
  } else {
    q = base[d - 64] * s + base[d] * c;
    k = base[128 + d - 64] * s + base[128 + d] * c;
  }
  const float v = base[256 + d];
  const size_t bh = (size_t)(b * HH + h);
  Qh[(bh * LL + l) * DD + d] = (_Float16)q;
  Kh[(bh * LL + l) * DD + d] = (_Float16)k;
  Vt[(bh * DD + d) * LL + l] = (_Float16)v;
}

// ---------------------------------------------------------------------------
// Kernel 3: flash attention. One (b,h) per blockIdx.x, 64-query tile per
// blockIdx.y; 128 threads = 4 waves, wave w owns query rows [qt*64+16w, +16).
// Shared 64x128 K tile and 128x64 V tile async-DMA'd into LDS per key step
// (all 4 waves consume them); S = Q K^T WMMA, register online softmax, P via
// wave-private LDS, O += P V WMMA. Writes attnO f32 + attnO16 f16.
// ---------------------------------------------------------------------------
__global__ __launch_bounds__(128) void k_flash_attn(
    const _Float16* __restrict__ Qh, const _Float16* __restrict__ Kh,
    const _Float16* __restrict__ Vt, float* __restrict__ attnO,
    _Float16* __restrict__ attnO16) {
  const int t    = threadIdx.x;
  const int lane = t & 31;
  const int wave = t >> 5;               // 0..3
  const int bh   = blockIdx.x;           // 0..63
  const int b    = bh >> 4, h = bh & 15;
  const int qbase = blockIdx.y * 64 + wave * 16;

  const _Float16* Qp = Qh + ((size_t)bh * LL + qbase) * DD;
  const _Float16* Kp = Kh + (size_t)bh * LL * DD;
  const _Float16* Vp = Vt + (size_t)bh * DD * LL;

  __shared__ alignas(16) _Float16 ldsK[64 * KP];     // 64 keys x 128 feats
  __shared__ alignas(16) _Float16 ldsV[128 * VP];    // 128 feats x 64 keys
  __shared__ alignas(16) _Float16 lds_p[4][16 * PLD];
  const unsigned ldsK0 = (unsigned)(size_t)(const void*)&ldsK[0];
  const unsigned ldsV0 = (unsigned)(size_t)(const void*)&ldsV[0];

  v16h qf[4];
#pragma unroll
  for (int kk = 0; kk < 4; ++kk) qf[kk] = load_frag_f16(Qp + kk * 32, DD, lane);

  v8f oacc[8] = {};
  float mstate[8], lstate[8];
#pragma unroll
  for (int r = 0; r < 8; ++r) { mstate[r] = -3.0e38f; lstate[r] = 0.0f; }

  const float scale = 0.08838834764831845f;   // 1/sqrt(128)

  for (int j = 0; j < LL / 64; ++j) {
    // ---- async-stage shared K (64x128) and V (128x64) tiles ----
#pragma unroll
    for (int i = 0; i < 8; ++i) {          // K: 1024 b128 / 128 thr
      const int idx = t + 128 * i;
      const int kr = idx >> 4, ks = idx & 15;
      async_load_b128(ldsK0 + (unsigned)(kr * KP + ks * 8) * 2,
                      Kp + (size_t)(j * 64 + kr) * DD + ks * 8);
    }
#pragma unroll
    for (int i = 0; i < 8; ++i) {          // V: 1024 b128 / 128 thr
      const int idx = t + 128 * i;
      const int vr = idx >> 3, vs = idx & 7;
      async_load_b128(ldsV0 + (unsigned)(vr * VP + vs * 8) * 2,
                      Vp + (size_t)vr * LL + j * 64 + vs * 8);
    }
    wait_async0();
    __syncthreads();
    // ---- S = Q K^T (4 n-tiles x 4 k-chunks) from LDS K tile ----
    v8f sacc[4] = {};
#pragma unroll
    for (int tt = 0; tt < 4; ++tt) {
      const _Float16* kb = &ldsK[(tt * 16) * KP];
#pragma unroll
      for (int kk = 0; kk < 4; ++kk) {
        v16h bf = load_frag_f16(kb + kk * 32, KP, lane);
        sacc[tt] = wmma_f32_f16(qf[kk], bf, sacc[tt]);
      }
    }
    // ---- online softmax (row = acc idx r + 8*(lane>=16); col = lane&15) ---
#pragma unroll
    for (int r = 0; r < 8; ++r) {
      float v0 = fmaxf(fmaxf(sacc[0][r], sacc[1][r]),
                       fmaxf(sacc[2][r], sacc[3][r])) * scale;
      v0 = fmaxf(v0, __shfl_xor(v0, 1, 32));
      v0 = fmaxf(v0, __shfl_xor(v0, 2, 32));
      v0 = fmaxf(v0, __shfl_xor(v0, 4, 32));
      v0 = fmaxf(v0, __shfl_xor(v0, 8, 32));
      const float mnew  = fmaxf(mstate[r], v0);
      const float alpha = __expf(mstate[r] - mnew);
      float sum = 0.0f;
#pragma unroll
      for (int tt = 0; tt < 4; ++tt) {
        const float p = __expf(sacc[tt][r] * scale - mnew);
        sacc[tt][r] = p;
        sum += p;
      }
      sum += __shfl_xor(sum, 1, 32);
      sum += __shfl_xor(sum, 2, 32);
      sum += __shfl_xor(sum, 4, 32);
      sum += __shfl_xor(sum, 8, 32);
      lstate[r] = lstate[r] * alpha + sum;
      mstate[r] = mnew;
#pragma unroll
      for (int n = 0; n < 8; ++n) oacc[n][r] *= alpha;
    }
    // ---- stage P (16x64 f16) into wave-private LDS, re-read as A frags ----
    {
      const int col = lane & 15, rowoff = (lane >> 4) << 3;
#pragma unroll
      for (int tt = 0; tt < 4; ++tt)
#pragma unroll
        for (int r = 0; r < 8; ++r)
          lds_p[wave][(rowoff + r) * PLD + tt * 16 + col] =
              (_Float16)sacc[tt][r];
    }
    asm volatile("s_wait_dscnt 0x0" ::: "memory");  // wave-private RAW via LDS
    v16h pf0 = load_frag_f16(&lds_p[wave][0], PLD, lane);
    v16h pf1 = load_frag_f16(&lds_p[wave][32], PLD, lane);
    // ---- O += P V from LDS V tile ----
#pragma unroll
    for (int n = 0; n < 8; ++n) {
      const _Float16* vb = &ldsV[(n * 16) * VP];
      v16h vf0 = load_frag_f16(vb, VP, lane);
      v16h vf1 = load_frag_f16(vb + 32, VP, lane);
      oacc[n] = wmma_f32_f16(pf0, vf0, oacc[n]);
      oacc[n] = wmma_f32_f16(pf1, vf1, oacc[n]);
    }
    __syncthreads();              // K/V tiles free for next key step
  }
  // ---- normalize and write attnO f32 + attnO16 f16, (B*L, H*D) ----
  const int col = lane & 15, rowoff = (lane >> 4) << 3;
#pragma unroll
  for (int n = 0; n < 8; ++n) {
#pragma unroll
    for (int r = 0; r < 8; ++r) {
      const int row = qbase + rowoff + r;
      const size_t o = ((size_t)(b * LL + row)) * DM + h * DD + n * 16 + col;
      const float val = oacc[n][r] / lstate[r];
      attnO[o]   = val;
      attnO16[o] = (_Float16)val;
    }
  }
}

// ---------------------------------------------------------------------------
// Kernel 5a: per-(b,l,h) entropy over the 128-dim head axis of attn output.
// ---------------------------------------------------------------------------
__global__ __launch_bounds__(256) void k_entropy_rows(
    const float* __restrict__ attnO, float* __restrict__ ent) {
  const int lane = threadIdx.x & 31;
  const int wave = threadIdx.x >> 5;
  const int row  = blockIdx.x * 8 + wave;        // 0 .. B*L*H-1
  const float* p = attnO + (size_t)(row >> 4) * DM + (size_t)(row & 15) * DD;
  float a[4];
#pragma unroll
  for (int i = 0; i < 4; ++i) a[i] = fabsf(p[lane * 4 + i]);
  float m = fmaxf(fmaxf(a[0], a[1]), fmaxf(a[2], a[3]));
#pragma unroll
  for (int msk = 1; msk < 32; msk <<= 1) m = fmaxf(m, __shfl_xor(m, msk, 32));
  float e = 0.0f, s1 = 0.0f, s2 = 0.0f;
#pragma unroll
  for (int i = 0; i < 4; ++i) {
    e  += __expf(a[i] - m);
    s1 += a[i];
    s2 += a[i] * a[i];
  }
#pragma unroll
  for (int msk = 1; msk < 32; msk <<= 1) {
    e  += __shfl_xor(e, msk, 32);
    s1 += __shfl_xor(s1, msk, 32);
    s2 += __shfl_xor(s2, msk, 32);
  }
  if (lane == 0) {
    const float lse = m + __logf(e);
    ent[row] = lse * s1 - s2;     // -sum((a-lse)*a)
  }
}

// Kernel 5b: deterministic mean of B*L*H entropies -> scalar tail of d_out.
__global__ __launch_bounds__(256) void k_reduce_mean(
    const float* __restrict__ ent, float* __restrict__ outp) {
  __shared__ float sm[256];
  float s = 0.0f;
  for (int i = threadIdx.x; i < BB * LL * HH; i += 256) s += ent[i];
  sm[threadIdx.x] = s;
  __syncthreads();
  for (int stride = 128; stride > 0; stride >>= 1) {
    if ((int)threadIdx.x < stride) sm[threadIdx.x] += sm[threadIdx.x + stride];
    __syncthreads();
  }
  if (threadIdx.x == 0) outp[0] = sm[0] / (float)(BB * LL * HH);
}

// ---------------------------------------------------------------------------
extern "C" void kernel_launch(void* const* d_in, const int* in_sizes, int n_in,
                              void* d_out, int out_size, void* d_ws,
                              size_t ws_size, hipStream_t stream) {
  const float* x    = (const float*)d_in[0];   // (B,L,DM)
  const float* cosv = (const float*)d_in[1];   // (L)
  const float* sinv = (const float*)d_in[2];   // (L)
  const float* wqkv = (const float*)d_in[3];   // (DM, H*3D)
  const float* bqkv = (const float*)d_in[4];   // (H*3D)
  const float* wout = (const float*)d_in[5];   // (H*D, DM)
  const float* bout = (const float*)d_in[6];   // (DM)
  float* out = (float*)d_out;                  // B*L*DM floats + 1 scalar

  char* ws = (char*)d_ws;
  size_t off = 0;
  float*    qkv32  = (float*)(ws + off);    off += (size_t)BB * LL * NQKV * 4;
  _Float16* wTq    = (_Float16*)(ws + off); off += (size_t)NQKV * KDIM * 2;
  _Float16* wTo    = (_Float16*)(ws + off); off += (size_t)DM * KDIM * 2;
  _Float16* xh     = (_Float16*)(ws + off); off += (size_t)BB * LL * DM * 2;
  _Float16* Qh     = (_Float16*)(ws + off); off += (size_t)BB * HH * LL * DD * 2;
  _Float16* Kh     = (_Float16*)(ws + off); off += (size_t)BB * HH * LL * DD * 2;
  _Float16* Vt     = (_Float16*)(ws + off); off += (size_t)BB * HH * DD * LL * 2;
  float*    attnO  = (float*)(ws + off);    off += (size_t)BB * LL * DM * 4;
  _Float16* attnO16= (_Float16*)(ws + off); off += (size_t)BB * LL * DM * 2;
  float*    ent    = (float*)(ws + off);    off += (size_t)BB * LL * HH * 4;
  (void)ws_size; (void)in_sizes; (void)n_in; (void)out_size;

  // 0) operand conversion: x -> f16; weights -> f16 transposed (N,K)
  k_convert_f16<<<(BB * LL * DM) / 256, 256, 0, stream>>>(x, xh, BB * LL * DM);
  k_transpose_f16<<<(KDIM * NQKV) / 256, 256, 0, stream>>>(wqkv, wTq, KDIM, NQKV);
  k_transpose_f16<<<(KDIM * DM) / 256, 256, 0, stream>>>(wout, wTo, KDIM, DM);
  // 1) fused QKV projection (+bias), double-buffered async-LDS WMMA GEMM
  k_gemm_f16<<<dim3((BB * LL) / 256, NQKV / 64), 256, 0, stream>>>(
      xh, wTq, bqkv, qkv32, NQKV);
  // 2) RoPE + pack to f16 (V transposed)
  k_rope_pack<<<(BB * HH * LL * DD) / 256, 256, 0, stream>>>(
      qkv32, cosv, sinv, Qh, Kh, Vt);
  // 3) flash attention -> attnO (f32) + attnO16 (f16)
  k_flash_attn<<<dim3(BB * HH, LL / 64), 128, 0, stream>>>(
      Qh, Kh, Vt, attnO, attnO16);
  // 4) output projection (+bias) -> d_out
  k_gemm_f16<<<dim3((BB * LL) / 256, DM / 64), 256, 0, stream>>>(
      attnO16, wTo, bout, out, DM);
  // 5) entropy diagnostic -> d_out tail scalar
  k_entropy_rows<<<(BB * LL * HH) / 8, 256, 0, stream>>>(attnO, ent);
  k_reduce_mean<<<1, 256, 0, stream>>>(ent, out + (size_t)BB * LL * DM);
}
